// BahdanauAttention_81965155876868
// MI455X (gfx1250) — matrix operand order repaired and checked by reading, not compiled, not run on hardware
//
#include <hip/hip_runtime.h>
#include <hip/hip_bf16.h>
#include <math.h>

// Problem constants (from reference): B=32, T=1024, D=2048, U=1024
#define BB 32
#define TT 1024
#define DD 2048
#define UU 1024

typedef __attribute__((ext_vector_type(16))) __bf16 v16bf;
typedef __attribute__((ext_vector_type(8)))  __bf16 v8bf;
typedef __attribute__((ext_vector_type(8)))  float  v8f;

static __device__ __forceinline__ float fast_tanh(float x) {
#if __has_builtin(__builtin_amdgcn_tanhf)
  return __builtin_amdgcn_tanhf(x);   // v_tanh_f32 on gfx1250
#else
  return tanhf(x);
#endif
}

// ---------------------------------------------------------------------------
// Kernel 1: W1 (D x U, f32) -> W1T (U x D, bf16), tiled transpose via LDS.
// ---------------------------------------------------------------------------
__global__ void __launch_bounds__(256)
w1_transpose_bf16_kernel(const float* __restrict__ W1, __bf16* __restrict__ W1T) {
  __shared__ float tile[32][33];
  const int tx = threadIdx.x & 31;
  const int ty = threadIdx.x >> 5;          // 0..7
  const int n0 = blockIdx.x * 32;           // U dimension
  const int k0 = blockIdx.y * 32;           // D dimension
#pragma unroll
  for (int i = 0; i < 4; ++i)
    tile[ty + i * 8][tx] = W1[(size_t)(k0 + ty + i * 8) * UU + (n0 + tx)];
  __syncthreads();
#pragma unroll
  for (int i = 0; i < 4; ++i)
    W1T[(size_t)(n0 + ty + i * 8) * DD + (k0 + tx)] = (__bf16)tile[tx][ty + i * 8];
}

// ---------------------------------------------------------------------------
// Kernel 2: qb1[b][n] = sum_k query[b][k] * W2[k][n] + b2[n] + b1[n]
// ---------------------------------------------------------------------------
__global__ void __launch_bounds__(256)
qproj_kernel(const float* __restrict__ query, const float* __restrict__ W2,
             const float* __restrict__ b1, const float* __restrict__ b2,
             float* __restrict__ qb1) {
  const int n = blockIdx.x * 256 + threadIdx.x;   // 0..1023
  const int b = blockIdx.y;                       // 0..31
  const float* qr = query + (size_t)b * DD;
  float acc = 0.f;
#pragma unroll 8
  for (int k = 0; k < DD; ++k)
    acc = fmaf(qr[k], W2[(size_t)k * UU + n], acc);
  qb1[b * UU + n] = acc + b1[n] + b2[n];
}

// ---------------------------------------------------------------------------
// Kernel 3 (dominant): fused  score = tanh(values@W1 + qb1) @ V + bv
// One WG (4 waves, 128 thr) owns 32 rows (one batch, 32 timesteps).
//  - values rows converted fp32->bf16 into LDS once (values read 1x from HBM)
//  - per wave: 2 M-blocks x 2 N-blocks = 4 WMMAs sharing A/B operands
//    (halves LDS traffic per FLOP vs 1 N-block)
// ---------------------------------------------------------------------------
#define LDA 2056                      // 2048 + 8 bf16 pad -> conflict-free b128
#define SCORE_SMEM (32 * LDA * 2 + 32 * 4)

__global__ void __launch_bounds__(128)
score_kernel(const float* __restrict__ values, const __bf16* __restrict__ W1T,
             const float* __restrict__ qb1, const float* __restrict__ Vw,
             const float* __restrict__ bv, float* __restrict__ score) {
  extern __shared__ char smem[];
  __bf16* As   = (__bf16*)smem;                     // 32 x LDA bf16
  float*  ssum = (float*)(smem + 32 * LDA * 2);     // 32 floats

  const int tid  = threadIdx.x;
  const int wave = tid >> 5;
  const int lane = tid & 31;
  const int m0   = blockIdx.x * 32;                 // global row base (b*T + t)
  const int b    = m0 >> 10;                        // / T

  // ---- stage A tile: 32 rows x 2048 cols, fp32 -> bf16 LDS (rows contiguous)
  const float* src = values + (size_t)m0 * DD;
  for (int idx = tid; idx < (32 * DD) / 4; idx += 128) {
    const int e   = idx * 4;
    const int row = e >> 11;
    const int col = e & (DD - 1);
    float4 f = *(const float4*)(src + e);
    __bf16* d = As + row * LDA + col;
    d[0] = (__bf16)f.x; d[1] = (__bf16)f.y; d[2] = (__bf16)f.z; d[3] = (__bf16)f.w;
  }
  if (tid < 32) ssum[tid] = 0.0f;
  __syncthreads();

  // Operand layouts used by this code (wave32; contiguous k, same for A and B):
  //  A 16x32 bf16: lane<16 -> K 0..15; lane>=16 -> K 16..31; M = lane%16
  //  B 32x16 bf16: lane = column N%16; lane<16 -> K 0..15, lane>=16 -> K 16..31
  //  C/D f32 16x16: VGPR r -> M = r + 8*(lane>=16); N = lane%16
  const int half  = lane >> 4;            // 0 / 1
  const int nlane = lane & 15;
  const int selA  = half * 16;
  const int selB  = half * 16;

  float p[16];
#pragma unroll
  for (int i = 0; i < 16; ++i) p[i] = 0.f;

  // each wave owns 32 columns of every 128-wide slab
  for (int n0 = 0; n0 < UU; n0 += 128) {
    const int na = n0 + wave * 32 + nlane;          // first 16-col block
    const int nb = na + 16;                         // second 16-col block
    const __bf16* Bp0 = W1T + (size_t)na * DD + selB;
    const __bf16* Bp1 = W1T + (size_t)nb * DD + selB;
    const __bf16* A0  = As + nlane * LDA + selA;        // rows 0..15
    const __bf16* A1  = As + (nlane + 16) * LDA + selA; // rows 16..31
    v8f acc00 = {}; v8f acc01 = {};
    v8f acc10 = {}; v8f acc11 = {};
#pragma unroll 2
    for (int k0 = 0; k0 < DD; k0 += 32) {
      v16bf b0  = *(const v16bf*)(Bp0 + k0);        // 2 x global b128 (L2 hit)
      v16bf b1v = *(const v16bf*)(Bp1 + k0);
      v8bf a0lo = *(const v8bf*)(A0 + k0);          // ds_load_b128
      v8bf a0hi = *(const v8bf*)(A0 + k0 + 8);
      v8bf a1lo = *(const v8bf*)(A1 + k0);
      v8bf a1hi = *(const v8bf*)(A1 + k0 + 8);
      v16bf a0 = __builtin_shufflevector(a0lo, a0hi,
                   0,1,2,3,4,5,6,7,8,9,10,11,12,13,14,15);
      v16bf a1 = __builtin_shufflevector(a1lo, a1hi,
                   0,1,2,3,4,5,6,7,8,9,10,11,12,13,14,15);
      acc00 = __builtin_amdgcn_wmma_f32_16x16x32_bf16(false, a0, false, b0,
                                                      (short)0, acc00, false, false);
      acc10 = __builtin_amdgcn_wmma_f32_16x16x32_bf16(false, a1, false, b0,
                                                      (short)0, acc10, false, false);
      acc01 = __builtin_amdgcn_wmma_f32_16x16x32_bf16(false, a0, false, b1v,
                                                      (short)0, acc01, false, false);
      acc11 = __builtin_amdgcn_wmma_f32_16x16x32_bf16(false, a1, false, b1v,
                                                      (short)0, acc11, false, false);
    }
    // epilogue: p[row] += tanh(kp + qb1[b][n]) * V[n] for both column blocks
    const float adda = qb1[b * UU + na];
    const float va   = Vw[na];
    const float addb = qb1[b * UU + nb];
    const float vb   = Vw[nb];
#pragma unroll
    for (int r = 0; r < 8; ++r) {
      p[r]     += fast_tanh(acc00[r] + adda) * va + fast_tanh(acc01[r] + addb) * vb;
      p[8 + r] += fast_tanh(acc10[r] + adda) * va + fast_tanh(acc11[r] + addb) * vb;
    }
  }

  // cross-lane / cross-wave reduction into 32 per-row scores
  const int rbase = half * 8;
#pragma unroll
  for (int r = 0; r < 8; ++r) {
    atomicAdd(&ssum[rbase + r],      p[r]);         // ds_add_f32
    atomicAdd(&ssum[16 + rbase + r], p[8 + r]);
  }
  __syncthreads();
  if (tid < 32) score[m0 + tid] = ssum[tid] + bv[0];
}

// ---------------------------------------------------------------------------
// Kernel 4: softmax over T per batch
// ---------------------------------------------------------------------------
__global__ void __launch_bounds__(256)
softmax_kernel(const float* __restrict__ score, float* __restrict__ wout) {
  __shared__ float red[256];
  const int b = blockIdx.x, tid = threadIdx.x;
  const float* s = score + (size_t)b * TT;
  float v[4];
  float lmax = -1e30f;
#pragma unroll
  for (int i = 0; i < 4; ++i) { v[i] = s[tid + i * 256]; lmax = fmaxf(lmax, v[i]); }
  red[tid] = lmax; __syncthreads();
  for (int off = 128; off > 0; off >>= 1) {
    if (tid < off) red[tid] = fmaxf(red[tid], red[tid + off]);
    __syncthreads();
  }
  const float m = red[0]; __syncthreads();
  float lsum = 0.f;
#pragma unroll
  for (int i = 0; i < 4; ++i) { v[i] = __expf(v[i] - m); lsum += v[i]; }
  red[tid] = lsum; __syncthreads();
  for (int off = 128; off > 0; off >>= 1) {
    if (tid < off) red[tid] += red[tid + off];
    __syncthreads();
  }
  const float inv = 1.0f / red[0];
#pragma unroll
  for (int i = 0; i < 4; ++i) wout[(size_t)b * TT + tid + i * 256] = v[i] * inv;
}

// ---------------------------------------------------------------------------
// Kernel 5: context[b][d] = sum_t w[b][t] * values[b][t][d]  (HBM-bound pass)
// ---------------------------------------------------------------------------
__global__ void __launch_bounds__(128)
context_kernel(const float* __restrict__ values, const float* __restrict__ w,
               float* __restrict__ ctx) {
  __shared__ float wl[TT];
  const int tid = threadIdx.x;
  const int b   = blockIdx.y;
  const int d   = blockIdx.x * 128 + tid;
  for (int i = tid; i < TT; i += 128) wl[i] = w[(size_t)b * TT + i];
  __syncthreads();
  const float* vp = values + (size_t)b * TT * DD + d;
  float acc = 0.f;
#pragma unroll 4
  for (int t = 0; t < TT; ++t) acc = fmaf(wl[t], vp[(size_t)t * DD], acc);
  ctx[(size_t)b * DD + d] = acc;
}

// ---------------------------------------------------------------------------
// Host launcher
// inputs: 0 query(32x2048) 1 values(32x1024x2048) 2 W1(2048x1024) 3 b1(1024)
//         4 W2(2048x1024) 5 b2(1024) 6 V(1024) 7 bv(1)
// output: context (32x2048) ++ attention_weights (32x1024)
// ---------------------------------------------------------------------------
extern "C" void kernel_launch(void* const* d_in, const int* in_sizes, int n_in,
                              void* d_out, int out_size, void* d_ws, size_t ws_size,
                              hipStream_t stream) {
  (void)in_sizes; (void)n_in; (void)out_size; (void)ws_size;
  const float* query  = (const float*)d_in[0];
  const float* values = (const float*)d_in[1];
  const float* W1     = (const float*)d_in[2];
  const float* b1     = (const float*)d_in[3];
  const float* W2     = (const float*)d_in[4];
  const float* b2     = (const float*)d_in[5];
  const float* Vw     = (const float*)d_in[6];
  const float* bv     = (const float*)d_in[7];

  // workspace layout
  char* ws = (char*)d_ws;
  __bf16* W1T   = (__bf16*)ws;                                      // 4 MB
  float*  qb1   = (float*)(ws + (size_t)UU * DD * 2);               // 128 KB
  float*  score = (float*)(ws + (size_t)UU * DD * 2 + BB * UU * 4); // 128 KB

  float* out_ctx = (float*)d_out;              // 32*2048
  float* out_w   = out_ctx + (size_t)BB * DD;  // 32*1024

  w1_transpose_bf16_kernel<<<dim3(UU / 32, DD / 32), 256, 0, stream>>>(W1, W1T);
  qproj_kernel<<<dim3(UU / 256, BB), 256, 0, stream>>>(query, W2, b1, b2, qb1);
  score_kernel<<<dim3((BB * TT) / 32), 128, SCORE_SMEM, stream>>>(
      values, W1T, qb1, Vw, bv, score);
  softmax_kernel<<<dim3(BB), 256, 0, stream>>>(score, out_w);
  context_kernel<<<dim3(DD / 128, BB), 128, 0, stream>>>(values, out_w, out_ctx);
}
